// MLPBlock_51445118272040
// MI455X (gfx1250) — matrix operand order repaired and checked by reading, not compiled, not run on hardware
//
#include <hip/hip_runtime.h>
#include <math.h>

// ---------------------------------------------------------------------------
// MoE MLP block (RMSNorm -> top2 router -> per-expert SwiGLU MLP -> combine)
// Routed-only compute: gather tokens per expert, grouped GEMMs on WMMA bf16.
// Weights pre-converted to bf16 once. GEMM staging via CDNA5 async global->LDS
// (ASYNCcnt), triple-buffered (2 stages in flight, s_wait_asynccnt <= 6).
// 64x128 block tile, 8 waves x (32x32, 4 accums), K-step 64.
// ---------------------------------------------------------------------------

typedef __bf16 bf16;
typedef __attribute__((ext_vector_type(16))) bf16 v16bf;
typedef __attribute__((ext_vector_type(8))) float v8f;

constexpr int kH = 1024;   // hidden
constexpr int kE = 8;      // experts
constexpr int kTopK = 2;   // experts per token
constexpr int kI = 1024;   // intermediate (mlp1 emits 2*kI interleaved)
constexpr int kN = 4096;   // tokens
constexpr float kAlpha = 1.702f;
constexpr float kLimit = 7.0f;
constexpr float kEps = 1e-5f;

constexpr int BM = 64;     // token tile
constexpr int BN = 128;    // output-column tile
constexpr int KT = 64;     // k per staging step (2 wmma k-depths)
constexpr int LDP = 72;    // LDS row stride (144B, 16B-aligned, bank-spread)
constexpr int NBUF = 3;    // staging buffers (2 async stages in flight)
constexpr int OPS_PER_STAGE = 6;  // async b128 copies per thread per stage

// ---- async global->LDS staging (CDNA5 GLOBAL_LOAD_ASYNC_TO_LDS_B128) ------
#if __has_builtin(__builtin_amdgcn_global_load_async_to_lds_b128)
#define ASYNC_LDS 1
#else
#define ASYNC_LDS 0
#pragma message("gfx1250 async-to-LDS builtin unavailable: sync LDS staging")
#endif

typedef int vint4 __attribute__((vector_size(16)));
typedef __attribute__((address_space(1))) vint4* gvec_p;   // global b128
typedef __attribute__((address_space(3))) vint4* lvec_p;   // LDS b128

__device__ __forceinline__ void cp16_to_lds(void* dst_lds, const void* src_g) {
#if ASYNC_LDS
  __builtin_amdgcn_global_load_async_to_lds_b128((gvec_p)src_g, (lvec_p)dst_lds,
                                                 0, 0);
#else
  *(uint4*)dst_lds = *(const uint4*)src_g;
#endif
}

// Wait until at most N async copies remain outstanding (in-order completion:
// guarantees the oldest stage has fully landed in LDS).
template <int N>
__device__ __forceinline__ void wait_async_stage() {
#if ASYNC_LDS
#if __has_builtin(__builtin_amdgcn_s_wait_asynccnt)
  __builtin_amdgcn_s_wait_asynccnt((short)N);
#else
  asm volatile("s_wait_asynccnt %0" ::"i"(N) : "memory");
#endif
#endif
}

// ---------------------------------------------------------------------------
// Kernel 0: fp32 -> bf16 weight conversion (one-time, streaming).
// ---------------------------------------------------------------------------
__global__ __launch_bounds__(256) void cvt_bf16_kernel(
    const float* __restrict__ src, bf16* __restrict__ dst, int n4) {
  const int i = blockIdx.x * 256 + threadIdx.x;
  if (i < n4) {
    const float4 f = ((const float4*)src)[i];
    union { bf16 h[4]; uint2 u; } tb;
    tb.h[0] = (bf16)f.x; tb.h[1] = (bf16)f.y;
    tb.h[2] = (bf16)f.z; tb.h[3] = (bf16)f.w;
    ((uint2*)dst)[i] = tb.u;
  }
}

// ---------------------------------------------------------------------------
// Kernel 1: RMSNorm + router. One block (256 thr) per token.
// ---------------------------------------------------------------------------
__global__ __launch_bounds__(256) void rms_gate_kernel(
    const float* __restrict__ x, const float* __restrict__ scale,
    const float* __restrict__ gw, const float* __restrict__ gb,
    float* __restrict__ out, bf16* __restrict__ T, int* __restrict__ cnt,
    int* __restrict__ ids, int* __restrict__ slots, float* __restrict__ wts) {
  const int tkn = blockIdx.x;
  const int tid = threadIdx.x;
  const int lane = tid & 31, wid = tid >> 5;

  const float4 v = ((const float4*)(x + (size_t)tkn * kH))[tid];
  float ss = v.x * v.x + v.y * v.y + v.z * v.z + v.w * v.w;
#pragma unroll
  for (int off = 16; off > 0; off >>= 1) ss += __shfl_xor(ss, off, 32);

  __shared__ float red[8];
  __shared__ float gsh[kE];
  if (lane == 0) red[wid] = ss;
  if (tid < kE) gsh[tid] = 0.0f;
  __syncthreads();

  float tot = 0.0f;
#pragma unroll
  for (int i = 0; i < 8; ++i) tot += red[i];
  const float rn = rsqrtf(tot * (1.0f / (float)kH) + kEps);

  const float4 sc = ((const float4*)scale)[tid];
  float tv[4];
  tv[0] = v.x * rn * sc.x;
  tv[1] = v.y * rn * sc.y;
  tv[2] = v.z * rn * sc.z;
  tv[3] = v.w * rn * sc.w;

  ((float4*)(out + (size_t)tkn * kH))[tid] = v;  // residual init

  union { bf16 h[4]; uint2 u; } tb;
#pragma unroll
  for (int q = 0; q < 4; ++q) tb.h[q] = (bf16)tv[q];
  *(uint2*)&T[(size_t)tkn * kH + tid * 4] = tb.u;

  float p[kE];
#pragma unroll
  for (int e = 0; e < kE; ++e) p[e] = 0.0f;
  const int hbase = tid * 4;
#pragma unroll
  for (int q = 0; q < 4; ++q) {
    const float* grow = gw + (size_t)(hbase + q) * kE;
#pragma unroll
    for (int e = 0; e < kE; ++e) p[e] += tv[q] * grow[e];
  }
#pragma unroll
  for (int e = 0; e < kE; ++e) atomicAdd(&gsh[e], p[e]);
  __syncthreads();

  if (tid == 0) {
    float g[kE];
#pragma unroll
    for (int e = 0; e < kE; ++e) g[e] = gsh[e] + gb[e];
    int b0 = 0;
#pragma unroll
    for (int e = 1; e < kE; ++e) if (g[e] > g[b0]) b0 = e;
    int b1 = (b0 == 0) ? 1 : 0;
#pragma unroll
    for (int e = 0; e < kE; ++e)
      if (e != b0 && g[e] > g[b1]) b1 = e;
    const float e1 = __expf(g[b1] - g[b0]);
    const float inv = 1.0f / (1.0f + e1);

    int p0 = atomicAdd(&cnt[b0], 1);
    ids[b0 * kN + p0] = tkn;
    slots[b0 * kN + p0] = tkn * kTopK + 0;
    wts[b0 * kN + p0] = inv;
    int p1 = atomicAdd(&cnt[b1], 1);
    ids[b1 * kN + p1] = tkn;
    slots[b1 * kN + p1] = tkn * kTopK + 1;
    wts[b1 * kN + p1] = e1 * inv;
  }
}

// ---------------------------------------------------------------------------
// WMMA fragment load (contiguous-k wave32 layout, same for A and B).
//  A 16x32: lane<16 row=lane  K0..15 contig;    lane>=16 row=lane-16 K16..31 contig
//  B 32x16: lane<16 col=lane  K0..15 contig;    lane>=16 col=lane-16 K16..31 contig
// ---------------------------------------------------------------------------
union FragU { v16bf v; uint4 u[2]; };

__device__ __forceinline__ v16bf load_a_frag(const bf16* p) {
  FragU a;
  a.u[0] = *(const uint4*)p;
  a.u[1] = *(const uint4*)(p + 8);
  return a.v;
}
__device__ __forceinline__ v16bf load_b_frag(const bf16* p) {
  FragU b;
  b.u[0] = *(const uint4*)p;
  b.u[1] = *(const uint4*)(p + 8);
  return b.v;
}
__device__ __forceinline__ v8f wmma_bf16(v16bf a, v16bf b, v8f c) {
  return __builtin_amdgcn_wmma_f32_16x16x32_bf16(false, a, false, b, (short)0,
                                                 c, false, false);
}

// ---------------------------------------------------------------------------
// Kernel 2: grouped GEMM1 + fused SwiGLU.  grid (2I/BN, N/BM, E)
// ---------------------------------------------------------------------------
__global__ __launch_bounds__(256) void gemm1_swiglu_kernel(
    const bf16* __restrict__ T, const bf16* __restrict__ W1,
    const float* __restrict__ B1, const int* __restrict__ cnt,
    const int* __restrict__ ids, const int* __restrict__ slots,
    bf16* __restrict__ H1) {
  __shared__ bf16 As[NBUF][BM][LDP];
  __shared__ bf16 Bs[NBUF][BN][LDP];
  __shared__ int sRid[BM], sSlot[BM];

  const int e = blockIdx.z;
  const int c = cnt[e];
  const int m0 = blockIdx.y * BM;
  if (m0 >= c) return;
  const int n0 = blockIdx.x * BN;  // column in [0, 2I)

  const int tid = threadIdx.x;
  if (tid < BM) {
    const int r = min(m0 + tid, c - 1);
    sRid[tid] = ids[e * kN + r];
    sSlot[tid] = slots[e * kN + r];
  }
  __syncthreads();

  // staging: A 64x64 bf16 (16/thr), B 128x64 bf16 (32/thr); hoisted pointers
  const int arow = tid >> 2, acol = (tid & 3) * 16;
  const int brow = tid >> 1, bcol = (tid & 1) * 32;
  const bf16* aptr = T + (size_t)sRid[arow] * kH + acol;
  const bf16* bptr = W1 + ((size_t)e * 2 * kI + n0 + brow) * kH + bcol;

  const int lane = tid & 31, wid = tid >> 5;
  const int wr = wid >> 2, wc = wid & 3;  // 2x4 wave grid, 32x32 per wave
  const int lh = lane & 15, hs = lane >> 4;
  const bf16* aFragBase = &As[0][wr * 32 + lh][hs * 16];
  const bf16* bFragBase = &Bs[0][wc * 32 + lh][hs * 16];
  constexpr int kBufA = BM * LDP;
  constexpr int kBufB = BN * LDP;

  v8f acc[2][2] = {{{}, {}}, {{}, {}}};

  auto stage = [&](int buf, int kt) {
    cp16_to_lds(&As[buf][arow][acol], aptr + kt);
    cp16_to_lds(&As[buf][arow][acol + 8], aptr + kt + 8);
#pragma unroll
    for (int q = 0; q < 4; ++q)
      cp16_to_lds(&Bs[buf][brow][bcol + q * 8], bptr + kt + q * 8);
  };

  constexpr int nk = kH / KT;
  stage(0, 0);
  stage(1, KT);
  int cbuf = 0;  // buffer consumed this iteration; stage goes to (cbuf+2)%3
#pragma unroll 1
  for (int s = 0; s < nk; ++s) {
    wait_async_stage<OPS_PER_STAGE>();  // oldest stage landed (ours)
    __syncthreads();                    // ... and everyone else's
    if (s + 2 < nk) {
      int sbuf = cbuf + 2;
      if (sbuf >= NBUF) sbuf -= NBUF;
      stage(sbuf, (s + 2) * KT);
    }
#pragma unroll
    for (int ks = 0; ks < KT; ks += 32) {
      const v16bf a0 = load_a_frag(aFragBase + cbuf * kBufA + ks);
      const v16bf a1 = load_a_frag(aFragBase + cbuf * kBufA + 16 * LDP + ks);
      const v16bf b0 = load_b_frag(bFragBase + cbuf * kBufB + ks);
      const v16bf b1 = load_b_frag(bFragBase + cbuf * kBufB + 16 * LDP + ks);
      acc[0][0] = wmma_bf16(a0, b0, acc[0][0]);
      acc[0][1] = wmma_bf16(a0, b1, acc[0][1]);
      acc[1][0] = wmma_bf16(a1, b0, acc[1][0]);
      acc[1][1] = wmma_bf16(a1, b1, acc[1][1]);
    }
    cbuf = (cbuf + 1 == NBUF) ? 0 : cbuf + 1;
  }

  // epilogue: +bias, pair even(gate)/odd(linear) cols via lane shfl, swiglu
  const bool evenLane = (lane & 1) == 0;
#pragma unroll
  for (int asel = 0; asel < 2; ++asel) {
#pragma unroll
    for (int bsel = 0; bsel < 2; ++bsel) {
      const int nglob = n0 + wc * 32 + bsel * 16 + lh;  // in [0,2I)
      const float bias = B1[e * 2 * kI + nglob];
      const int pcol = (n0 >> 1) + wc * 16 + bsel * 8 + (lh >> 1);
#pragma unroll
      for (int j = 0; j < 8; ++j) {
        const float v = acc[asel][bsel][j] + bias;
        const float pv = __shfl_xor(v, 1, 32);  // partner column
        if (evenLane) {
          const float xg = fminf(v, kLimit);
          const float xl = fminf(fmaxf(pv, -kLimit), kLimit);
          const float sg = 1.0f / (1.0f + __expf(-kAlpha * xg));
          const float h = xg * sg * (xl + 1.0f);
          const int mrow = wr * 32 + asel * 16 + hs * 8 + j;
          if (m0 + mrow < c)
            H1[(size_t)sSlot[mrow] * kI + pcol] = (bf16)h;
        }
      }
    }
  }
}

// ---------------------------------------------------------------------------
// Kernel 3: grouped GEMM2 + weighted scatter-add.  grid (H/BN, N/BM, E)
// ---------------------------------------------------------------------------
__global__ __launch_bounds__(256) void gemm2_scatter_kernel(
    const bf16* __restrict__ H1, const bf16* __restrict__ W2,
    const float* __restrict__ B2, const int* __restrict__ cnt,
    const int* __restrict__ ids, const int* __restrict__ slots,
    const float* __restrict__ wts, float* __restrict__ out) {
  __shared__ bf16 As[NBUF][BM][LDP];
  __shared__ bf16 Bs[NBUF][BN][LDP];
  __shared__ int sRid[BM], sSlot[BM];
  __shared__ float sWt[BM];

  const int e = blockIdx.z;
  const int c = cnt[e];
  const int m0 = blockIdx.y * BM;
  if (m0 >= c) return;
  const int n0 = blockIdx.x * BN;  // column in [0, H)

  const int tid = threadIdx.x;
  if (tid < BM) {
    const int r = min(m0 + tid, c - 1);
    sRid[tid] = ids[e * kN + r];
    sSlot[tid] = slots[e * kN + r];
    sWt[tid] = wts[e * kN + r];
  }
  __syncthreads();

  const int arow = tid >> 2, acol = (tid & 3) * 16;
  const int brow = tid >> 1, bcol = (tid & 1) * 32;
  const bf16* aptr = H1 + (size_t)sSlot[arow] * kI + acol;
  const bf16* bptr = W2 + ((size_t)e * kH + n0 + brow) * kI + bcol;

  const int lane = tid & 31, wid = tid >> 5;
  const int wr = wid >> 2, wc = wid & 3;
  const int lh = lane & 15, hs = lane >> 4;
  const bf16* aFragBase = &As[0][wr * 32 + lh][hs * 16];
  const bf16* bFragBase = &Bs[0][wc * 32 + lh][hs * 16];
  constexpr int kBufA = BM * LDP;
  constexpr int kBufB = BN * LDP;

  v8f acc[2][2] = {{{}, {}}, {{}, {}}};

  auto stage = [&](int buf, int kt) {
    cp16_to_lds(&As[buf][arow][acol], aptr + kt);
    cp16_to_lds(&As[buf][arow][acol + 8], aptr + kt + 8);
#pragma unroll
    for (int q = 0; q < 4; ++q)
      cp16_to_lds(&Bs[buf][brow][bcol + q * 8], bptr + kt + q * 8);
  };

  constexpr int nk = kI / KT;
  stage(0, 0);
  stage(1, KT);
  int cbuf = 0;
#pragma unroll 1
  for (int s = 0; s < nk; ++s) {
    wait_async_stage<OPS_PER_STAGE>();
    __syncthreads();
    if (s + 2 < nk) {
      int sbuf = cbuf + 2;
      if (sbuf >= NBUF) sbuf -= NBUF;
      stage(sbuf, (s + 2) * KT);
    }
#pragma unroll
    for (int ks = 0; ks < KT; ks += 32) {
      const v16bf a0 = load_a_frag(aFragBase + cbuf * kBufA + ks);
      const v16bf a1 = load_a_frag(aFragBase + cbuf * kBufA + 16 * LDP + ks);
      const v16bf b0 = load_b_frag(bFragBase + cbuf * kBufB + ks);
      const v16bf b1 = load_b_frag(bFragBase + cbuf * kBufB + 16 * LDP + ks);
      acc[0][0] = wmma_bf16(a0, b0, acc[0][0]);
      acc[0][1] = wmma_bf16(a0, b1, acc[0][1]);
      acc[1][0] = wmma_bf16(a1, b0, acc[1][0]);
      acc[1][1] = wmma_bf16(a1, b1, acc[1][1]);
    }
    cbuf = (cbuf + 1 == NBUF) ? 0 : cbuf + 1;
  }

#pragma unroll
  for (int asel = 0; asel < 2; ++asel) {
#pragma unroll
    for (int bsel = 0; bsel < 2; ++bsel) {
      const int hcol = n0 + wc * 32 + bsel * 16 + lh;
      const float bias = B2[e * kH + hcol];
#pragma unroll
      for (int j = 0; j < 8; ++j) {
        const int mrow = wr * 32 + asel * 16 + hs * 8 + j;
        if (m0 + mrow < c) {
          const float v = sWt[mrow] * (acc[asel][bsel][j] + bias);
          atomicAdd(out + (size_t)sRid[mrow] * kH + hcol, v);
        }
      }
    }
  }
}

// ---------------------------------------------------------------------------
// Launch. ws = t | cnt | ids | slots | wts | h1 | W1bf | W2bf  (~73 MB)
// ---------------------------------------------------------------------------
extern "C" void kernel_launch(void* const* d_in, const int* in_sizes, int n_in,
                              void* d_out, int out_size, void* d_ws,
                              size_t ws_size, hipStream_t stream) {
  const float* x = (const float*)d_in[0];
  const float* norm_scale = (const float*)d_in[1];
  const float* gate_w = (const float*)d_in[2];
  const float* gate_b = (const float*)d_in[3];
  const float* mlp1_w = (const float*)d_in[4];
  const float* mlp1_b = (const float*)d_in[5];
  const float* mlp2_w = (const float*)d_in[6];
  const float* mlp2_b = (const float*)d_in[7];
  float* out = (float*)d_out;

  char* ws = (char*)d_ws;
  size_t off = 0;
  bf16* T = (bf16*)(ws + off);     off += (size_t)kN * kH * sizeof(bf16);
  int* cnt = (int*)(ws + off);     off += 256;  // padded
  int* ids = (int*)(ws + off);     off += (size_t)kE * kN * sizeof(int);
  int* slots = (int*)(ws + off);   off += (size_t)kE * kN * sizeof(int);
  float* wts = (float*)(ws + off); off += (size_t)kE * kN * sizeof(float);
  bf16* H1 = (bf16*)(ws + off);    off += (size_t)kN * kTopK * kI * sizeof(bf16);
  bf16* W1b = (bf16*)(ws + off);   off += (size_t)kE * 2 * kI * kH * sizeof(bf16);
  bf16* W2b = (bf16*)(ws + off);   // E*H*I bf16

  (void)hipMemsetAsync(cnt, 0, kE * sizeof(int), stream);

  const int n1 = kE * 2 * kI * kH / 4;  // float4 chunks in W1
  const int n2 = kE * kH * kI / 4;
  cvt_bf16_kernel<<<(n1 + 255) / 256, 256, 0, stream>>>(mlp1_w, W1b, n1);
  cvt_bf16_kernel<<<(n2 + 255) / 256, 256, 0, stream>>>(mlp2_w, W2b, n2);

  rms_gate_kernel<<<kN, 256, 0, stream>>>(x, norm_scale, gate_w, gate_b, out,
                                          T, cnt, ids, slots, wts);
  gemm1_swiglu_kernel<<<dim3(2 * kI / BN, kN / BM, kE), 256, 0, stream>>>(
      T, W1b, mlp1_b, cnt, ids, slots, H1);
  gemm2_scatter_kernel<<<dim3(kH / BN, kN / BM, kE), 256, 0, stream>>>(
      H1, W2b, mlp2_b, cnt, ids, slots, wts, out);
}
